// LatentAttentionPooling_65395172048974
// MI455X (gfx1250) — matrix operand order repaired and checked by reading, not compiled, not run on hardware
//
#include <hip/hip_runtime.h>
#include <hip/hip_bf16.h>

// ---------------------------------------------------------------------------
// LatentAttentionPooling on MI455X (gfx1250, wave32, WMMA bf16)
//
// Folded algorithm (mathematically identical to reference):
//   qt[r=h*8+k][c] = scale * sum_j Wk[h*128+j][c] * q[k][h*128+j]
//   scores[b,r,s]  = qt[r] . hidden[b,s]  + scale*(q[h,k].bk_h)      (GEMM #1)
//   p = softmax_s(masked scores)          (store unnormalized exp, keep sum)
//   u[b,r,:]       = sum_s p[b,r,s] * hidden[b,s,:]                  (GEMM #2)
//   ctx[b,k,dp]    = Wv[dp] . (u[b,r]/sum) + bv[dp]    (r = (dp>>7)*8 + k)
//   pooled[b,k,:]  = ctx[b,k] @ Wo^T + bo ; LayerNorm ; mean over k
// ---------------------------------------------------------------------------

typedef __attribute__((ext_vector_type(16))) __bf16          v16bf;
typedef __attribute__((ext_vector_type(8)))  float           v8f;
typedef __attribute__((ext_vector_type(16))) unsigned short  v16u;
typedef __attribute__((ext_vector_type(8)))  unsigned short  v8u;
typedef __attribute__((ext_vector_type(8)))  unsigned int    v8ui;
typedef __attribute__((ext_vector_type(4)))  int             v4i;

#define B_      8
#define S_      8192
#define D_      1024
#define H_      8
#define HD_     128
#define K_LAT   8
#define R_      64           // H_*K_LAT flattened (h-major)
#define SCALE_  0.0883883476483184f   // 1/sqrt(128)

#if __has_builtin(__builtin_amdgcn_global_load_async_to_lds_b128)
#define HAVE_ASYNC_LDS 1
typedef __attribute__((address_space(1))) v4i* g_v4i;   // global (device) AS
typedef __attribute__((address_space(3))) v4i* l_v4i;   // LDS (shared) AS
#if __has_builtin(__builtin_amdgcn_s_wait_asynccnt)
#define WAIT_ASYNC(n) __builtin_amdgcn_s_wait_asynccnt(n)
#else
#define WAIT_ASYNC(n) asm volatile("s_wait_asynccnt %0" :: "i"(n) : "memory")
#endif
#else
#define HAVE_ASYNC_LDS 0
#define WAIT_ASYNC(n)
#endif

__device__ __forceinline__ unsigned short f2bf(float f) {
    unsigned int u = __float_as_uint(f) + 0x8000u;     // round-to-nearest
    return (unsigned short)(u >> 16);
}

// pack bf16(lo), bf16(hi) into one dword: 2 adds + 1 v_perm_b32
#if __has_builtin(__builtin_amdgcn_perm)
__device__ __forceinline__ unsigned int f2bf2(float lo, float hi) {
    unsigned int a = __float_as_uint(lo) + 0x8000u;
    unsigned int b = __float_as_uint(hi) + 0x8000u;
    return __builtin_amdgcn_perm(b, a, 0x07060302u);   // {b.hi16, a.hi16}
}
#else
__device__ __forceinline__ unsigned int f2bf2(float lo, float hi) {
    return (unsigned int)f2bf(lo) | ((unsigned int)f2bf(hi) << 16);
}
#endif

// ---------------------------------------------------------------------------
// K0a: qf[k][j] = bq[j] + sum_c lat[k][c] * Wq[j][c]          (8x1024 outputs)
// ---------------------------------------------------------------------------
__global__ void k_proj_q(const float* __restrict__ lat, const float* __restrict__ W,
                         const float* __restrict__ bias, float* __restrict__ qf) {
    int tid = blockIdx.x * 256 + threadIdx.x;          // 8192 threads
    int k = tid >> 10, j = tid & 1023;
    const float* lr = lat + k * D_;
    const float* wr = W + (size_t)j * D_;              // Wq rows 0..1023
    float acc = 0.f;
    for (int c = 0; c < D_; ++c) acc += lr[c] * wr[c];
    qf[tid] = acc + bias[j];
}

// ---------------------------------------------------------------------------
// K0b: qt[r][c] folded + packed straight into WMMA B-operand lane layout.
//   pack[(c/32)*4 + (r/16)][lane*16 + t], lane = ((c%32>>3)&1)*16 + r%16,
//   t = c%8 + (c%32>=16 ? 8:0).  (B 32x16 bf16: lanes 0-15 hold K={0..7,16..23}
//   of col n=lane, lanes 16-31 hold K={8..15,24..31} of col n=lane-16.)
// Also sbias[r] = scale * (q[h,k] . bk_h)   computed by the c==0 thread.
// ---------------------------------------------------------------------------
__global__ void k_fold_qt(const float* __restrict__ Wk, const float* __restrict__ bk,
                          const float* __restrict__ qf, unsigned short* __restrict__ qtp,
                          float* __restrict__ sbias) {
    int tid = blockIdx.x * 256 + threadIdx.x;          // 65536 threads
    int r = tid >> 10, c = tid & 1023;
    int h = r >> 3, k = r & 7;
    const float* qrow = qf + k * D_ + h * HD_;
    float acc = 0.f;
    for (int j = 0; j < HD_; ++j) acc += Wk[(size_t)(h * HD_ + j) * D_ + c] * qrow[j];
    acc *= SCALE_;
    int ck = c >> 5, within = c & 31;
    int lane = (((within >> 3) & 1) << 4) + (r & 15);
    int t = (within & 7) + ((within >> 4) << 3);
    qtp[((size_t)(ck * 4 + (r >> 4))) * 512 + lane * 16 + t] = f2bf(acc);
    if (c == 0) {
        float sb = 0.f;
        for (int j = 0; j < HD_; ++j) sb += qrow[j] * bk[h * HD_ + j];
        sbias[r] = sb * SCALE_;
    }
}

// ---------------------------------------------------------------------------
// K1: scores GEMM.  M = 65536 token rows, N = 64 (r), K = 1024.
// Block = 4 waves; wave w owns its OWN 16-row M-tile and iterates all 4
// r-tiles with 4 accumulators (A converted once per 4 WMMAs).
// A (hidden f32 -> bf16) per 16-bit A layout:
//   lanes 0-15 : M=lane,    K = kc+{0..7, 16..23}
//   lanes 16-31: M=lane-16, K = kc+{8..15, 24..31}
// D stored transposed: scoresT[b][r][s] (softmax rows contiguous).
// ---------------------------------------------------------------------------
__global__ __launch_bounds__(128) void
k_scores(const float* __restrict__ hs, const unsigned short* __restrict__ qtp,
         const float* __restrict__ sbias, float* __restrict__ scoresT) {
    const int lane = threadIdx.x & 31;
    const int wave = threadIdx.x >> 5;
    const long long srow0 = ((long long)blockIdx.x * 4 + wave) * 16;   // token row
    const int b  = (int)(srow0 >> 13);
    const int s0 = (int)(srow0 & (S_ - 1));
    const int m    = lane & 15;
    const int half = lane >> 4;
    const int koff = half * 8;
    const float* arow = hs + (size_t)(srow0 + m) * D_ + koff;
    v8f acc[4];
    #pragma unroll
    for (int rt = 0; rt < 4; ++rt) acc[rt] = (v8f){0.f,0.f,0.f,0.f,0.f,0.f,0.f,0.f};
    for (int kc = 0; kc < D_; kc += 32) {
        const float* ap = arow + kc;
        __builtin_prefetch(ap + 128, 0, 1);
        float4 f0 = *(const float4*)(ap);
        float4 f1 = *(const float4*)(ap + 4);
        float4 f2 = *(const float4*)(ap + 16);
        float4 f3 = *(const float4*)(ap + 20);
        v8ui aw;
        aw[0] = f2bf2(f0.x, f0.y); aw[1] = f2bf2(f0.z, f0.w);
        aw[2] = f2bf2(f1.x, f1.y); aw[3] = f2bf2(f1.z, f1.w);
        aw[4] = f2bf2(f2.x, f2.y); aw[5] = f2bf2(f2.z, f2.w);
        aw[6] = f2bf2(f3.x, f3.y); aw[7] = f2bf2(f3.z, f3.w);
        v16bf a = __builtin_bit_cast(v16bf, aw);
        const unsigned short* bbase = qtp + ((size_t)(kc >> 5) * 4) * 512 + lane * 16;
        #pragma unroll
        for (int rt = 0; rt < 4; ++rt) {
            v16u bv = *(const v16u*)(bbase + rt * 512);
            acc[rt] = __builtin_amdgcn_wmma_f32_16x16x32_bf16(
                          false, a, false, __builtin_bit_cast(v16bf, bv),
                          (short)0, acc[rt], false, false);
        }
    }
    // D layout: lanes 0-15: (M=g, N=lane); lanes 16-31: (M=g+8, N=lane-16)
    const int n = lane & 15;
    #pragma unroll
    for (int rt = 0; rt < 4; ++rt) {
        const int r = rt * 16 + n;
        const float bias = sbias[r];
        float* drow = scoresT + (((size_t)b * R_ + r) << 13) + s0 + (half << 3);
        float4 o0 = make_float4(acc[rt][0] + bias, acc[rt][1] + bias,
                                acc[rt][2] + bias, acc[rt][3] + bias);
        float4 o1 = make_float4(acc[rt][4] + bias, acc[rt][5] + bias,
                                acc[rt][6] + bias, acc[rt][7] + bias);
        *(float4*)(drow)     = o0;
        *(float4*)(drow + 4) = o1;
    }
}

// ---------------------------------------------------------------------------
// K2: masked softmax over s for one (b,r) row; stores UNNORMALIZED exp (bf16)
//     and the row sum (normalization folded into the Wv epilogue).
// ---------------------------------------------------------------------------
__global__ void k_softmax(const float* __restrict__ scoresT, const int* __restrict__ am,
                          unsigned short* __restrict__ P, float* __restrict__ sumexp) {
    __shared__ float red[256];
    const int row = blockIdx.x;                 // b*64 + r
    const int b = row >> 6;
    const int tid = threadIdx.x;
    const float* base = scoresT + ((size_t)row << 13);
    const int* mrow = am + b * S_;
    float m = -3.0e38f;
    for (int s = tid; s < S_; s += 256)
        if (mrow[s] != 0) m = fmaxf(m, base[s]);
    red[tid] = m; __syncthreads();
    for (int w = 128; w > 0; w >>= 1) {
        if (tid < w) red[tid] = fmaxf(red[tid], red[tid + w]);
        __syncthreads();
    }
    m = red[0]; __syncthreads();
    float sum = 0.f;
    unsigned short* prow = P + ((size_t)row << 13);
    for (int s = tid; s < S_; s += 256) {
        float e = (mrow[s] != 0) ? __expf(base[s] - m) : 0.f;
        prow[s] = f2bf(e);
        sum += e;
    }
    red[tid] = sum; __syncthreads();
    for (int w = 128; w > 0; w >>= 1) {
        if (tid < w) red[tid] += red[tid + w];
        __syncthreads();
    }
    if (tid == 0) sumexp[row] = red[0];
}

// ---------------------------------------------------------------------------
// K3: u GEMM.  Per b: M = 64 (r), N = 1024 (c), K = 8192 (s).
// A = P (bf16, layout-native loads from global), B = hidden staged through
// a DOUBLE-BUFFERED async-to-LDS pipeline, then read column-wise + cvt.
// Grid 512 blocks x 128: blk -> (b, rt, cg); wave owns ct = cg*4+wave.
// LDS: 2 x [s=32][c=64] f32 tiles (16 KB); lane column reads hit distinct
// banks.  s_wait_asynccnt 4 retires the current tile while the next tile's
// 4 async b128 ops stay in flight (per-wave in-order ASYNCcnt).
// ---------------------------------------------------------------------------
__global__ __launch_bounds__(128) void
k_ugemm(const unsigned short* __restrict__ P, const float* __restrict__ hs,
        float* __restrict__ u) {
    __shared__ float tile[2][32 * 64];
    const int tid  = threadIdx.x;
    const int lane = tid & 31;
    const int wave = tid >> 5;
    const int blk  = blockIdx.x;
    const int b  = blk >> 6;
    const int rt = (blk >> 4) & 3;
    const int cg = blk & 15;                    // 64-column group
    const int c0 = cg * 64;
    const int m    = lane & 15;
    const int half = lane >> 4;
    const int r = rt * 16 + m;
    const unsigned short* prow = P + (((size_t)b * R_ + r) << 13) + (half << 3);
    const float* gbase = hs + ((size_t)b << 13) * D_ + c0;
    const int cl = wave * 16 + m;               // local column for B operand

    auto stage = [&](int kc, int buf) {
        #pragma unroll
        for (int q = 0; q < 4; ++q) {
            int ch = tid + q * 128;             // 512 chunks of 16B
            int srow = ch >> 4;
            int coff = (ch & 15) * 4;
            const float* src = gbase + (size_t)(kc + srow) * D_ + coff;
            float* dst = &tile[buf][srow * 64 + coff];
#if HAVE_ASYNC_LDS
            __builtin_amdgcn_global_load_async_to_lds_b128(
                (g_v4i)(v4i*)(const_cast<float*>(src)), (l_v4i)(v4i*)dst, 0, 0);
#else
            *(float4*)dst = *(const float4*)src;
#endif
        }
    };

    v8f acc = {0.f, 0.f, 0.f, 0.f, 0.f, 0.f, 0.f, 0.f};
    stage(0, 0);                                // prologue: tile 0 in flight
    for (int it = 0; it < S_ / 32; ++it) {
        const int kc = it * 32;
        // ---- A operand: probabilities, already bf16 in native layout ----
        v8u p0 = *(const v8u*)(prow + kc);          // K = koff..koff+7
        v8u p1 = *(const v8u*)(prow + kc + 16);     // K = koff+16..koff+23
        // ---- issue next tile, then retire current (keep next in flight) ----
        if (it + 1 < S_ / 32) {
            stage(kc + 32, (it + 1) & 1);
            WAIT_ASYNC(4);
        } else {
            WAIT_ASYNC(0);
        }
        __syncthreads();
        v16u av;
        #pragma unroll
        for (int t = 0; t < 8; ++t) { av[t] = p0[t]; av[8 + t] = p1[t]; }
        // ---- B operand: column cl of LDS tile, rows half*8+{0..7, 16..23} ----
        const float* col = &tile[it & 1][(half * 8) * 64 + cl];
        v8ui bw;
        #pragma unroll
        for (int t = 0; t < 8; ++t)
            bw[t] = f2bf2(col[(2 * t + (t >> 2) * 8) * 64], col[(2 * t + 1 + (t >> 2) * 8) * 64]);
        acc = __builtin_amdgcn_wmma_f32_16x16x32_bf16(
                  false, __builtin_bit_cast(v16bf, av),
                  false, __builtin_bit_cast(v16bf, bw),
                  (short)0, acc, false, false);
        __syncthreads();
    }
    const int ct = cg * 4 + wave;
    const int c2 = ct * 16 + (lane & 15);
    float* ubase = u + (((size_t)b * R_ + rt * 16 + 8 * half)) * D_ + c2;
    #pragma unroll
    for (int g = 0; g < 8; ++g) ubase[(size_t)g * D_] = acc[g];
}

// ---------------------------------------------------------------------------
// K4a: ctx[b,k,dp] = Wv[dp].(u[b,r]/sum) + bv[dp],  r=(dp>>7)*8+k
// ---------------------------------------------------------------------------
__global__ void k_ctx(const float* __restrict__ u, const float* __restrict__ sumexp,
                      const float* __restrict__ Wfull, const float* __restrict__ bfull,
                      float* __restrict__ ctx) {
    int tid = blockIdx.x * 256 + threadIdx.x;         // 65536
    int b = tid >> 13, rem = tid & 8191;
    int k = rem >> 10, dp = rem & 1023;
    int h = dp >> 7;
    int r = h * 8 + k;
    float inv = 1.f / sumexp[b * R_ + r];
    const float* wr = Wfull + (size_t)(2 * D_ + dp) * D_;     // Wv row
    const float* ur = u + ((size_t)b * R_ + r) * D_;
    float acc = 0.f;
    for (int c = 0; c < D_; ++c) acc += wr[c] * ur[c];
    ctx[tid] = acc * inv + bfull[2 * D_ + dp];
}

// ---------------------------------------------------------------------------
// K4b: pooled[b,k,j] = ctx[b,k] . Wo[j] + bo[j]
// ---------------------------------------------------------------------------
__global__ void k_outproj(const float* __restrict__ ctx, const float* __restrict__ Wo,
                          const float* __restrict__ bo, float* __restrict__ pooled) {
    int tid = blockIdx.x * 256 + threadIdx.x;         // 65536
    int j = tid & 1023;
    const float* cr = ctx + (size_t)(tid >> 10) * D_;
    const float* wr = Wo + (size_t)j * D_;
    float acc = 0.f;
    for (int c = 0; c < D_; ++c) acc += cr[c] * wr[c];
    pooled[tid] = acc + bo[j];
}

// ---------------------------------------------------------------------------
// K4c: LayerNorm per (b,k) then mean over k.  One block per b.
// ---------------------------------------------------------------------------
__global__ void k_finish(const float* __restrict__ pooled, const float* __restrict__ g,
                         const float* __restrict__ be, float* __restrict__ out) {
    __shared__ float red[256];
    const int b = blockIdx.x, tid = threadIdx.x;
    float oacc[4] = {0.f, 0.f, 0.f, 0.f};
    for (int k = 0; k < K_LAT; ++k) {
        const float* row = pooled + ((size_t)(b * K_LAT + k)) * D_;
        float x[4], s = 0.f;
        #pragma unroll
        for (int t = 0; t < 4; ++t) { x[t] = row[tid + 256 * t]; s += x[t]; }
        red[tid] = s; __syncthreads();
        for (int w = 128; w > 0; w >>= 1) {
            if (tid < w) red[tid] += red[tid + w];
            __syncthreads();
        }
        float mu = red[0] * (1.f / D_); __syncthreads();
        float vs = 0.f;
        #pragma unroll
        for (int t = 0; t < 4; ++t) { float d = x[t] - mu; vs += d * d; }
        red[tid] = vs; __syncthreads();
        for (int w = 128; w > 0; w >>= 1) {
            if (tid < w) red[tid] += red[tid + w];
            __syncthreads();
        }
        float rinv = rsqrtf(red[0] * (1.f / D_) + 1e-5f); __syncthreads();
        #pragma unroll
        for (int t = 0; t < 4; ++t) {
            int j = tid + 256 * t;
            oacc[t] += (x[t] - mu) * rinv * g[j] + be[j];
        }
    }
    #pragma unroll
    for (int t = 0; t < 4; ++t) out[b * D_ + tid + 256 * t] = oacc[t] * 0.125f;
}

// ---------------------------------------------------------------------------
extern "C" void kernel_launch(void* const* d_in, const int* in_sizes, int n_in,
                              void* d_out, int out_size, void* d_ws, size_t ws_size,
                              hipStream_t stream) {
    (void)in_sizes; (void)n_in; (void)out_size; (void)ws_size;
    const float* hs   = (const float*)d_in[0];
    const float* lat  = (const float*)d_in[1];
    const float* Wip  = (const float*)d_in[2];
    const float* bip  = (const float*)d_in[3];
    const float* Wo   = (const float*)d_in[4];
    const float* bo   = (const float*)d_in[5];
    const float* lng  = (const float*)d_in[6];
    const float* lnb  = (const float*)d_in[7];
    const int*   am   = (const int*)d_in[8];
    float* out = (float*)d_out;

    char* ws = (char*)d_ws;
    float*          qf      = (float*)(ws + 0);                    //  32 KB
    float*          sbias   = (float*)(ws + (32 << 10));           // 256 B
    unsigned short* qtp     = (unsigned short*)(ws + (36 << 10));  // 128 KB
    float*          scoresT = (float*)(ws + (256 << 10));          //  16 MB
    float*          sumexp  = (float*)(ws + (256 << 10) + ((size_t)16 << 20)); // 2 KB
    unsigned short* P       = (unsigned short*)(ws + ((size_t)18 << 20));      // 8 MB
    float*          u       = (float*)(ws + ((size_t)26 << 20));   //  2 MB
    float*          ctx     = (float*)(ws + ((size_t)28 << 20));   // 256 KB
    float*          pooled  = (float*)(ws + ((size_t)28 << 20) + (256 << 10)); // 256 KB

    const float* Wk = Wip + (size_t)D_ * D_;   // rows 1024..2047
    const float* bk = bip + D_;

    k_proj_q  <<<32,   256, 0, stream>>>(lat, Wip, bip, qf);
    k_fold_qt <<<256,  256, 0, stream>>>(Wk, bk, qf, qtp, sbias);
    k_scores  <<<1024, 128, 0, stream>>>(hs, qtp, sbias, scoresT);
    k_softmax <<<512,  256, 0, stream>>>(scoresT, am, P, sumexp);
    k_ugemm   <<<512,  128, 0, stream>>>(P, hs, u);
    k_ctx     <<<256,  256, 0, stream>>>(u, sumexp, Wip, bip, ctx);
    k_outproj <<<256,  256, 0, stream>>>(ctx, Wo, bo, pooled);
    k_finish  <<<8,    256, 0, stream>>>(pooled, lng, lnb, out);
}
